// TransitionModel_76982993813766
// MI455X (gfx1250) — matrix operand order repaired and checked by reading, not compiled, not run on hardware
//
#include <hip/hip_runtime.h>

#define NB 256     // batch (M of the GEMM)
#define NN 1024    // state dim (N and K of the GEMM)

typedef __attribute__((ext_vector_type(16))) _Float16 v16h;
typedef __attribute__((ext_vector_type(8)))  float    v8f;
typedef __attribute__((ext_vector_type(2)))  _Float16 h2;

// ---------------------------------------------------------------------------
// Kernel 1: colLSE[j] = logsumexp over k of W[k,j]   (coalesced across j)
// ---------------------------------------------------------------------------
__global__ void col_lse_kernel(const float* __restrict__ W,
                               float* __restrict__ colLSE) {
    int j = blockIdx.x * blockDim.x + threadIdx.x;   // 0..NN-1
    float m = -3.402823466e38f;
    for (int k = 0; k < NN; ++k)
        m = fmaxf(m, W[(size_t)k * NN + j]);
    float s = 0.0f;
    for (int k = 0; k < NN; ++k)
        s += __expf(W[(size_t)k * NN + j] - m);
    colLSE[j] = m + __logf(s);
}

// ---------------------------------------------------------------------------
// Kernel 2: Qt[j,k] = (f16) exp(W[k,j] - colLSE[j])  -- LDS-tiled transpose
//           (K-contiguous storage so B-fragment K-pairs are one b32 load)
// ---------------------------------------------------------------------------
__global__ void exp_transpose_kernel(const float* __restrict__ W,
                                     const float* __restrict__ colLSE,
                                     _Float16* __restrict__ Qt) {
    __shared__ _Float16 tile[32][34];   // 17-dword row stride, coprime w/ 64 banks
    int j0 = blockIdx.x * 32;
    int k0 = blockIdx.y * 32;
    int c  = threadIdx.x & 31;
    int r0 = threadIdx.x >> 5;
    float lse = colLSE[j0 + c];
#pragma unroll
    for (int i = 0; i < 4; ++i) {
        int r = r0 + i * 8;
        tile[r][c] = (_Float16)__expf(W[(size_t)(k0 + r) * NN + (j0 + c)] - lse);
    }
    __syncthreads();
#pragma unroll
    for (int i = 0; i < 4; ++i) {
        int jr = r0 + i * 8;
        Qt[(size_t)(j0 + jr) * NN + (k0 + c)] = tile[c][jr];
    }
}

// ---------------------------------------------------------------------------
// Kernel 3: rowM[b] = max_k log_alpha[b,k];  P[b,k] = (f16) exp(a - rowM)
// ---------------------------------------------------------------------------
__global__ void row_softmax_kernel(const float* __restrict__ A,
                                   float* __restrict__ rowM,
                                   _Float16* __restrict__ P) {
    __shared__ float red[256];
    int b = blockIdx.x;
    int t = threadIdx.x;
    float m = -3.402823466e38f;
#pragma unroll
    for (int i = 0; i < 4; ++i)
        m = fmaxf(m, A[(size_t)b * NN + t + i * 256]);
    red[t] = m;
    __syncthreads();
    for (int off = 128; off > 0; off >>= 1) {
        if (t < off) red[t] = fmaxf(red[t], red[t + off]);
        __syncthreads();
    }
    float M = red[0];
#pragma unroll
    for (int i = 0; i < 4; ++i) {
        int k = t + i * 256;
        P[(size_t)b * NN + k] = (_Float16)__expf(A[(size_t)b * NN + k] - M);
    }
    if (t == 0) rowM[b] = M;
}

// ---------------------------------------------------------------------------
// Fragment loaders (gfx1250 16-bit WMMA VGPR layouts)
// ---------------------------------------------------------------------------
__device__ __forceinline__ v16h load_a_frag(const _Float16* __restrict__ Prow,
                                            int kk, int grp) {
    // A 16x32: VGPR v holds K = (v>=4)*16 + grp*8 + 2*(v&3) + {0, 1}, row M = lane%16
    v16h a;
#pragma unroll
    for (int v = 0; v < 8; ++v) {
        int k = kk + ((v >> 2) << 4) + grp * 8 + 2 * (v & 3);
        h2 p = *(const h2*)(Prow + k);
        a[2 * v]     = p[0];
        a[2 * v + 1] = p[1];
    }
    return a;
}

__device__ __forceinline__ v16h load_b_frag(const _Float16* __restrict__ Qcol,
                                            int kk, int grp) {
    // B 32x16: VGPR v holds rows K = (v>=4)*16 + grp*8 + 2*(v&3) + {0, 1}, col N = lane%16
    v16h b;
#pragma unroll
    for (int v = 0; v < 8; ++v) {
        h2 q = *(const h2*)(Qcol + kk + ((v >> 2) << 4) + grp * 8 + 2 * (v & 3));
        b[2 * v]     = q[0];
        b[2 * v + 1] = q[1];
    }
    return b;
}

// ---------------------------------------------------------------------------
// Kernel 4: out[b,j] = rowM[b] + log( (P @ Q)[b,j] )
//   One wave32 computes a 32x64 output strip: 2 M-tiles x 4 N-tiles,
//   8 independent v_wmma_f32_16x16x32_f16 chains, software-pipelined K-loop
//   with ping-pong register buffers (all indices compile-time constant).
// ---------------------------------------------------------------------------
__global__ void __launch_bounds__(256) lse_mm_kernel(
        const _Float16* __restrict__ P,    // [NB, NN] row-major
        const _Float16* __restrict__ Qt,   // [NN(j), NN(k)] : K contiguous
        const float*    __restrict__ rowM, // [NB]
        float*          __restrict__ out)  // [NB, NN]
{
    const int lane = threadIdx.x & 31;
    const int wave = threadIdx.x >> 5;
    const int gw   = blockIdx.x * 8 + wave;   // 128 waves total
    const int mg   = gw & 7;                  // 8 groups of 32 rows  (M = 256)
    const int ng   = gw >> 3;                 // 16 groups of 64 cols (N = 1024)
    const int grp  = lane >> 4;
    const int ln   = lane & 15;

    const _Float16* Prow[2];
    Prow[0] = P + (size_t)(mg * 32 + ln) * NN;
    Prow[1] = Prow[0] + (size_t)16 * NN;

    const _Float16* Qcol[4];
#pragma unroll
    for (int t = 0; t < 4; ++t)
        Qcol[t] = Qt + (size_t)(ng * 64 + t * 16 + ln) * NN;

    v8f acc[2][4] = {};

    // ---- software pipeline: ping-pong fragment buffers ----
    v16h A0[2], A1[2], B0[4], B1[4];
#pragma unroll
    for (int m = 0; m < 2; ++m) A0[m] = load_a_frag(Prow[m], 0, grp);
#pragma unroll
    for (int t = 0; t < 4; ++t) B0[t] = load_b_frag(Qcol[t], 0, grp);

    int kk = 0;
    for (; kk < NN - 64; kk += 64) {
        // prefetch kk+32 while computing kk
#pragma unroll
        for (int m = 0; m < 2; ++m) A1[m] = load_a_frag(Prow[m], kk + 32, grp);
#pragma unroll
        for (int t = 0; t < 4; ++t) B1[t] = load_b_frag(Qcol[t], kk + 32, grp);
#pragma unroll
        for (int m = 0; m < 2; ++m)
#pragma unroll
            for (int t = 0; t < 4; ++t)
                acc[m][t] = __builtin_amdgcn_wmma_f32_16x16x32_f16(
                    false, A0[m], false, B0[t], (short)0, acc[m][t], false, false);

        // prefetch kk+64 while computing kk+32
#pragma unroll
        for (int m = 0; m < 2; ++m) A0[m] = load_a_frag(Prow[m], kk + 64, grp);
#pragma unroll
        for (int t = 0; t < 4; ++t) B0[t] = load_b_frag(Qcol[t], kk + 64, grp);
#pragma unroll
        for (int m = 0; m < 2; ++m)
#pragma unroll
            for (int t = 0; t < 4; ++t)
                acc[m][t] = __builtin_amdgcn_wmma_f32_16x16x32_f16(
                    false, A1[m], false, B1[t], (short)0, acc[m][t], false, false);
    }

    // epilogue: kk == NN-64; A0/B0 hold step kk, fetch kk+32, drain both
#pragma unroll
    for (int m = 0; m < 2; ++m) A1[m] = load_a_frag(Prow[m], kk + 32, grp);
#pragma unroll
    for (int t = 0; t < 4; ++t) B1[t] = load_b_frag(Qcol[t], kk + 32, grp);
#pragma unroll
    for (int m = 0; m < 2; ++m)
#pragma unroll
        for (int t = 0; t < 4; ++t)
            acc[m][t] = __builtin_amdgcn_wmma_f32_16x16x32_f16(
                false, A0[m], false, B0[t], (short)0, acc[m][t], false, false);
#pragma unroll
    for (int m = 0; m < 2; ++m)
#pragma unroll
        for (int t = 0; t < 4; ++t)
            acc[m][t] = __builtin_amdgcn_wmma_f32_16x16x32_f16(
                false, A1[m], false, B1[t], (short)0, acc[m][t], false, false);

    // ---- epilogue: C/D layout VGPR r -> M = r (lanes 0-15) / 8+r (16-31) ----
#pragma unroll
    for (int m = 0; m < 2; ++m)
#pragma unroll
        for (int t = 0; t < 4; ++t) {
            int j = ng * 64 + t * 16 + ln;
#pragma unroll
            for (int r = 0; r < 8; ++r) {
                int b = mg * 32 + m * 16 + grp * 8 + r;
                out[(size_t)b * NN + j] = rowM[b] + __logf(acc[m][t][r]);
            }
        }
}

// ---------------------------------------------------------------------------
extern "C" void kernel_launch(void* const* d_in, const int* in_sizes, int n_in,
                              void* d_out, int out_size, void* d_ws, size_t ws_size,
                              hipStream_t stream) {
    const float* log_alpha = (const float*)d_in[0];   // [256, 1024] f32
    const float* W         = (const float*)d_in[1];   // [1024, 1024] f32
    // d_in[2] (use_max) is unused by the reference.
    float* out = (float*)d_out;                       // [256, 1024] f32

    // Workspace layout (~2.6 MB): colLSE | rowM | P (f16) | Qt (f16)
    char* ws = (char*)d_ws;
    float*    colLSE = (float*)ws;                                   //   4 KB
    float*    rowM   = (float*)(ws + 4096);                          //   1 KB
    _Float16* P      = (_Float16*)(ws + 8192);                       // 512 KB
    _Float16* Qt     = (_Float16*)(ws + 8192 + (size_t)NB * NN * 2); //   2 MB
    (void)in_sizes; (void)n_in; (void)out_size; (void)ws_size;

    col_lse_kernel<<<NN / 256, 256, 0, stream>>>(W, colLSE);

    dim3 g2(NN / 32, NN / 32);
    exp_transpose_kernel<<<g2, 256, 0, stream>>>(W, colLSE, Qt);

    row_softmax_kernel<<<NB, 256, 0, stream>>>(log_alpha, rowM, P);

    // 128 waves, 8 waves (256 threads) per block -> 16 blocks
    lse_mm_kernel<<<16, 256, 0, stream>>>(P, Qt, rowM, out);
}
